// GCNSample_58789512348190
// MI455X (gfx1250) — matrix operand order repaired and checked by reading, not compiled, not run on hardware
//
#include <hip/hip_runtime.h>
#include <hip/hip_bf16.h>
#include <stdint.h>

// ---------------------------------------------------------------------------
// CDNA5 WMMA types (wave32):
//   v_wmma_f32_16x16x32_bf16: A,B = 16 bf16/lane (8 VGPRs), C/D = 8 f32/lane
// ---------------------------------------------------------------------------
typedef __attribute__((ext_vector_type(16))) __bf16       v16bf;
typedef __attribute__((ext_vector_type(8)))  float        v8f;
typedef __attribute__((ext_vector_type(4)))  unsigned int u32x4;

// Exact operand type of the async-to-LDS builtin (leaked by the round-2
// diagnostic): GCC-style int4 vector, addrspace(1) src / addrspace(3) dst.
typedef int i32x4 __attribute__((__vector_size__(16)));
typedef __attribute__((address_space(1))) i32x4* g_b128_ptr;
typedef __attribute__((address_space(3))) i32x4* l_b128_ptr;

#define NNODES 100000
#define NEDGES 1600000
#define NFEAT  512
#define NOUT   128   // both hidden dims are 128

#ifndef __has_builtin
#define __has_builtin(x) 0
#endif

#if __has_builtin(__builtin_amdgcn_global_load_async_to_lds_b128)
#define HAVE_ASYNC_LDS 1
#else
#define HAVE_ASYNC_LDS 0
#endif

// 16-byte global -> LDS copy; async (ASYNCcnt path) when the toolchain has it.
__device__ __forceinline__ void copy16_g2l(const void* g, void* l) {
#if HAVE_ASYNC_LDS
  __builtin_amdgcn_global_load_async_to_lds_b128(
      (g_b128_ptr)(uintptr_t)g,
      (l_b128_ptr)(uint32_t)(uintptr_t)l, 0, 0);
#else
  *reinterpret_cast<u32x4*>(l) = *reinterpret_cast<const u32x4*>(g);
#endif
}

__device__ __forceinline__ void wait_async_copies() {
#if HAVE_ASYNC_LDS
#if __has_builtin(__builtin_amdgcn_s_wait_asynccnt)
  __builtin_amdgcn_s_wait_asynccnt(0);
#else
  asm volatile("s_wait_asynccnt 0x0" ::: "memory");
#endif
#endif
}

// ---------------------------------------------------------------------------
// fp32 -> bf16 one-time conversion (weights)
// ---------------------------------------------------------------------------
__global__ void __launch_bounds__(256)
cvt_f32_to_bf16(const float* __restrict__ in, __bf16* __restrict__ out, int n) {
  const int i = blockIdx.x * 256 + threadIdx.x;
  if (i < n) out[i] = (__bf16)in[i];
}

// ---------------------------------------------------------------------------
// GEMM: Z[M x 128] = A[M x K] @ W[K x 128]
//   A: fp32 in global, converted to bf16 while staging 32-row tile into LDS
//   W: pre-converted bf16, staged per 128-deep K-chunk via async-to-LDS
//   8 waves / block; wave w owns columns [16w,16w+16); 2 accumulators per
//   wave (rows 0-15 / 16-31 of the tile) -> 2 WMMAs per B fragment.
// ---------------------------------------------------------------------------
template <int K, bool RELU_A>
__global__ void __launch_bounds__(256)
gemm_bf16_wmma(const float* __restrict__ A, const __bf16* __restrict__ Wbf,
               float* __restrict__ Z) {
  __shared__ __align__(16) __bf16 As[32][K];        // 32 rows of A (bf16)
  __shared__ __align__(16) __bf16 Ws[128 * NOUT];   // 128-deep K-chunk of W

  const int tid = threadIdx.x;
  const int m0  = blockIdx.x * 32;

  // ---- Stage A tile: fp32 global -> bf16 LDS (b128 loads, fused ReLU) ----
  constexpr int K4 = K / 4;
  for (int i = tid; i < 32 * K4; i += 256) {
    const int r = i / K4;
    const int c = (i - r * K4) * 4;
    const float4 v =
        *reinterpret_cast<const float4*>(&A[(size_t)(m0 + r) * K + c]);
    float x0 = v.x, x1 = v.y, x2 = v.z, x3 = v.w;
    if (RELU_A) {
      x0 = fmaxf(x0, 0.0f); x1 = fmaxf(x1, 0.0f);
      x2 = fmaxf(x2, 0.0f); x3 = fmaxf(x3, 0.0f);
    }
    As[r][c + 0] = (__bf16)x0;
    As[r][c + 1] = (__bf16)x1;
    As[r][c + 2] = (__bf16)x2;
    As[r][c + 3] = (__bf16)x3;
  }

  const int wave = tid >> 5;
  const int lane = tid & 31;
  const int m    = lane & 15;          // A row within half-tile
  const int grp  = (lane >> 4) << 3;   // K-group: 0 (lanes 0-15) / 8 (16-31)
  const int ncol = wave * 16;          // output column base for this wave

  v8f acc0 = {};
  v8f acc1 = {};

  for (int kc = 0; kc < K; kc += 128) {
    __syncthreads();  // A tile ready (1st iter) / Ws no longer read (later)

    // ---- Stage 128x128 bf16 W chunk into LDS (async b128 copies) ----
    const __bf16* wsrc = Wbf + (size_t)kc * NOUT;
    for (int i = tid; i < (128 * NOUT) / 8; i += 256) {
      copy16_g2l(wsrc + (size_t)i * 8, Ws + i * 8);
    }
    wait_async_copies();
    __syncthreads();

#pragma unroll
    for (int kt = 0; kt < 4; ++kt) {
      // B fragment (32x16 bf16): lane = column ncol+(lane&15), elements = K rows (table order)
      v16bf b;
      {
        const __bf16* pb = &Ws[(kt * 32 + grp) * NOUT + ncol + (lane & 15)];
#pragma unroll
        for (int j = 0; j < 8; ++j) {
          b[j] = pb[j * NOUT];  b[j + 8] = pb[(j + 16) * NOUT];
        }
      }
      // A fragments (16x32 bf16 ISA layout): two contiguous 16B LDS reads
      v16bf a0, a1;
      {
        const int cbase = kc + kt * 32 + grp;
        const __bf16* p0 = &As[m][cbase];
        const __bf16* q0 = &As[m + 16][cbase];
#pragma unroll
        for (int j = 0; j < 8; ++j) {
          a0[j] = p0[j];  a0[j + 8] = p0[j + 16];
          a1[j] = q0[j];  a1[j + 8] = q0[j + 16];
        }
      }
      acc0 = __builtin_amdgcn_wmma_f32_16x16x32_bf16(
          false, a0, false, b, (short)0, acc0, false, false);
      acc1 = __builtin_amdgcn_wmma_f32_16x16x32_bf16(
          false, a1, false, b, (short)0, acc1, false, false);
    }
  }

  // C/D layout: VGPR r -> row (grp + r), column ncol + (lane & 15)
#pragma unroll
  for (int r = 0; r < 8; ++r) {
    const int col = ncol + (lane & 15);
    Z[(size_t)(m0 + grp + r) * NOUT + col]      = acc0[r];
    Z[(size_t)(m0 + 16 + grp + r) * NOUT + col] = acc1[r];
  }
}

// ---------------------------------------------------------------------------
// out[i*128 + f] = bias[f]   (accumulator init; deterministic each call)
// ---------------------------------------------------------------------------
__global__ void __launch_bounds__(256)
init_bias(float* __restrict__ out, const float* __restrict__ bias, int n) {
  const int i = blockIdx.x * 256 + threadIdx.x;
  if (i < n) out[i] = bias[i & (NOUT - 1)];
}

// ---------------------------------------------------------------------------
// COO SpMM: out[src[e]] += val[e] * dense[dst[e]]   (feature dim 128)
// One wave32 per edge; 4 floats/lane: b128 gather + 4x global_atomic_add_f32.
// ---------------------------------------------------------------------------
__global__ void __launch_bounds__(256)
spmm_scatter(const int* __restrict__ src, const int* __restrict__ dst,
             const float* __restrict__ val, const float* __restrict__ dense,
             float* __restrict__ out, int nedges) {
  const long long t = (long long)blockIdx.x * 256 + threadIdx.x;
  const long long e = t >> 5;
  const int lane    = (int)(t & 31);
  if (e >= nedges) return;

  const int s   = src[e];
  const int d   = dst[e];
  const float v = val[e];

  const float4 g =
      *reinterpret_cast<const float4*>(dense + (size_t)d * NOUT + lane * 4);
  float* op = out + (size_t)s * NOUT + lane * 4;
  atomicAdd(op + 0, g.x * v);
  atomicAdd(op + 1, g.y * v);
  atomicAdd(op + 2, g.z * v);
  atomicAdd(op + 3, g.w * v);
}

// ---------------------------------------------------------------------------
// In-place ReLU (final activation on d_out)
// ---------------------------------------------------------------------------
__global__ void __launch_bounds__(256)
relu_inplace(float* __restrict__ x, int n) {
  const int i = blockIdx.x * 256 + threadIdx.x;
  if (i < n) x[i] = fmaxf(x[i], 0.0f);
}

// ---------------------------------------------------------------------------
// Host launch
// ---------------------------------------------------------------------------
extern "C" void kernel_launch(void* const* d_in, const int* in_sizes, int n_in,
                              void* d_out, int out_size, void* d_ws,
                              size_t ws_size, hipStream_t stream) {
  const float* x        = (const float*)d_in[0];  // [100000, 512]
  const float* adj_val  = (const float*)d_in[1];  // [1.6M]
  const float* w1       = (const float*)d_in[2];  // [512, 128]
  const float* b1       = (const float*)d_in[3];  // [128]
  const float* w2       = (const float*)d_in[4];  // [128, 128]
  const float* b2       = (const float*)d_in[5];  // [128]
  const int*   edge_src = (const int*)d_in[6];    // [1.6M]
  const int*   edge_dst = (const int*)d_in[7];    // [1.6M]
  float*       out      = (float*)d_out;          // [100000, 128]

  // Workspace layout (all 256B-aligned offsets):
  //   z    : 100000*128 f32  (pre-aggregation activations, reused both layers)
  //   h    : 100000*128 f32  (layer-1 post-SpMM accumulator)
  //   w1bf : 512*128 bf16
  //   w2bf : 128*128 bf16
  float*  z    = (float*)d_ws;
  float*  h    = z + (size_t)NNODES * NOUT;
  __bf16* w1bf = (__bf16*)(h + (size_t)NNODES * NOUT);
  __bf16* w2bf = w1bf + (size_t)NFEAT * NOUT;

  const int nelem       = NNODES * NOUT;
  const int elem_blocks = (nelem + 255) / 256;
  const int spmm_blocks = (int)(((long long)NEDGES * 32 + 255) / 256);

  // One-time weight conversion to bf16
  cvt_f32_to_bf16<<<(NFEAT * NOUT + 255) / 256, 256, 0, stream>>>(
      w1, w1bf, NFEAT * NOUT);
  cvt_f32_to_bf16<<<(NOUT * NOUT + 255) / 256, 256, 0, stream>>>(
      w2, w2bf, NOUT * NOUT);

  // Layer 1: z = x @ w1 (bf16 WMMA, async W staging)
  gemm_bf16_wmma<NFEAT, false><<<NNODES / 32, 256, 0, stream>>>(x, w1bf, z);
  // h = b1; h[src] += val * z[dst]
  init_bias<<<elem_blocks, 256, 0, stream>>>(h, b1, nelem);
  spmm_scatter<<<spmm_blocks, 256, 0, stream>>>(edge_src, edge_dst, adj_val,
                                                z, h, NEDGES);
  // Layer 2: z = relu(h) @ w2 (ReLU fused into A staging)
  gemm_bf16_wmma<NOUT, true><<<NNODES / 32, 256, 0, stream>>>(h, w2bf, z);
  // out = b2; out[src] += val * z[dst]; out = relu(out)
  init_bias<<<elem_blocks, 256, 0, stream>>>(out, b2, nelem);
  spmm_scatter<<<spmm_blocks, 256, 0, stream>>>(edge_src, edge_dst, adj_val,
                                                z, out, NEDGES);
  relu_inplace<<<elem_blocks, 256, 0, stream>>>(out, nelem);
}
